// JointClassifier_36730560315475
// MI455X (gfx1250) — matrix operand order repaired and checked by reading, not compiled, not run on hardware
//
#include <hip/hip_runtime.h>
#include <math.h>

typedef __bf16 bf16;
typedef bf16  v16bf __attribute__((ext_vector_type(16)));
typedef bf16  v8bf  __attribute__((ext_vector_type(8)));
typedef float v8f   __attribute__((ext_vector_type(8)));
typedef float v4f   __attribute__((ext_vector_type(4)));

#define B_    8
#define L_    128
#define EMB_  300
#define POS_  30
#define IN_   330
#define INP_  352          // IN_ padded to a multiple of 32 for guard-free WMMA
#define H_    256
#define H2_   512
#define G4_   1024
#define G3_   1536
#define NBIO_ 9
#define NLAB_ 10
#define NEG_  (-100000000000.0f)

__device__ __forceinline__ float sigmoidf_(float x){ return 1.0f/(1.0f+expf(-x)); }

// CDNA5 bf16 16x32 A-fragment: elem j -> K = k0 + j + 8*((j>>3)+half)
//  => two contiguous 8-half (16B) runs at k0+8*half and k0+16+8*half
__device__ __forceinline__ v16bf loadA_frag(const bf16* p, int k0, int half){
  v8bf lo = *(const v8bf*)(p + k0 + 8*half);
  v8bf hi = *(const v8bf*)(p + k0 + 16 + 8*half);
  return __builtin_shufflevector(lo, hi, 0,1,2,3,4,5,6,7,8,9,10,11,12,13,14,15);
}
// CDNA5 bf16 32x16 B-fragment: elem j -> K = k0 + j + 8*((j>>3)+half)
//  => two contiguous 8-half (16B) runs at k0+8*half and k0+16+8*half
__device__ __forceinline__ v16bf loadB_frag(const bf16* p, int k0, int half){
  v8bf lo = *(const v8bf*)(p + k0 + 8*half);
  v8bf hi = *(const v8bf*)(p + k0 + 16 + 8*half);
  return __builtin_shufflevector(lo, hi, 0,1,2,3,4,5,6,7,8,9,10,11,12,13,14,15);
}

// ---------------------------------------------------------------- embedding
__global__ void embed_kernel(const int* __restrict__ tok, const int* __restrict__ pos,
                             const float* __restrict__ emb, const float* __restrict__ pemb,
                             float* __restrict__ x){
  int bl = blockIdx.x;
  for (int c = threadIdx.x; c < IN_; c += blockDim.x){
    float v;
    if (c < EMB_) v = emb[(size_t)tok[bl]*EMB_ + c];
    else          v = pemb[(size_t)pos[bl]*POS_ + (c-EMB_)];
    x[(size_t)bl*IN_ + c] = v;
  }
}

// ----------------------------- f32 -> bf16 pack with row/col zero padding
__global__ void pack_bf16_kernel(bf16* __restrict__ dst, const float* __restrict__ src,
                                 int dstRows, int dstCols, int srcRows, int srcCols,
                                 int srcStride, int srcOff){
  int i = blockIdx.x*blockDim.x + threadIdx.x;
  if (i >= dstRows*dstCols) return;
  int r = i / dstCols, c = i % dstCols;
  dst[i] = (r < srcRows && c < srcCols) ? (bf16)src[(size_t)r*srcStride + srcOff + c]
                                        : (bf16)0.0f;
}

// ------------------- guard-free WMMA GEMM: C = A*W^T (+bias), 32x32 per wave
// Requires: M,N multiples of 32; K (=lda=ldw) multiple of 32 (padded packs).
__global__ __launch_bounds__(32)
void gemm_wmma_kernel(const bf16* __restrict__ A, int lda,
                      const bf16* __restrict__ W, int ldw,
                      const float* __restrict__ bias,
                      float* __restrict__ C, int ldc, int K){
  int n0 = blockIdx.x*32, m0 = blockIdx.y*32;
  int lane = threadIdx.x & 31, half = lane >> 4, row = lane & 15;
  v8f acc00={0,0,0,0,0,0,0,0}, acc01=acc00, acc10=acc00, acc11=acc00;
  const bf16* A0 = A + (size_t)(m0 + row)*lda;
  const bf16* A1 = A + (size_t)(m0 + 16 + row)*lda;
  const bf16* W0 = W + (size_t)(n0 + row)*ldw;
  const bf16* W1 = W + (size_t)(n0 + 16 + row)*ldw;
  for (int k0 = 0; k0 < K; k0 += 32){
    v16bf a0 = loadA_frag(A0, k0, half);
    v16bf a1 = loadA_frag(A1, k0, half);
    v16bf b0 = loadB_frag(W0, k0, half);
    v16bf b1 = loadB_frag(W1, k0, half);
    acc00 = __builtin_amdgcn_wmma_f32_16x16x32_bf16(false, a0, false, b0, (short)0, acc00, false, false);
    acc01 = __builtin_amdgcn_wmma_f32_16x16x32_bf16(false, a0, false, b1, (short)0, acc01, false, false);
    acc10 = __builtin_amdgcn_wmma_f32_16x16x32_bf16(false, a1, false, b0, (short)0, acc10, false, false);
    acc11 = __builtin_amdgcn_wmma_f32_16x16x32_bf16(false, a1, false, b1, (short)0, acc11, false, false);
  }
  float bn0 = bias ? bias[n0 + row]      : 0.0f;
  float bn1 = bias ? bias[n0 + 16 + row] : 0.0f;
#pragma unroll
  for (int r = 0; r < 8; ++r){
    int m  = m0 + r + 8*half;
    C[(size_t)m*ldc + n0 + row]            = acc00[r] + bn0;
    C[(size_t)m*ldc + n0 + 16 + row]       = acc01[r] + bn1;
    C[(size_t)(m+16)*ldc + n0 + row]       = acc10[r] + bn0;
    C[(size_t)(m+16)*ldc + n0 + 16 + row]  = acc11[r] + bn1;
  }
}

// ----------------------------------- persistent LSTM recurrence (1 blk/dir)
__global__ __launch_bounds__(256)
void lstm_kernel(const float* __restrict__ Gx_f, const float* __restrict__ Gx_b,
                 const bf16* __restrict__ Whh_f, const bf16* __restrict__ Whh_b,
                 const float* __restrict__ mask,
                 float* __restrict__ hf, float* __restrict__ hb){
  int dir = blockIdx.x;
  const float* Gx  = dir ? Gx_b  : Gx_f;
  const bf16*  Whh = dir ? Whh_b : Whh_f;
  float*       out = dir ? hb    : hf;
  __shared__ float h_s[B_*H_];
  __shared__ float c_s[B_*H_];
  __shared__ alignas(16) bf16 hbf_s[16*H_];      // rows 8..15 stay zero (M pad)
  __shared__ float g_s[B_*G4_];
  int tid = threadIdx.x;
  for (int i = tid; i < B_*H_; i += 256){ h_s[i]=0.0f; c_s[i]=0.0f; }
  for (int i = tid; i < 16*H_; i += 256) hbf_s[i] = (bf16)0.0f;
  __syncthreads();
  int wave = tid >> 5, lane = tid & 31, half = lane >> 4, row = lane & 15;
  const bf16* hrow = hbf_s + row*H_;
  for (int t = 0; t < L_; ++t){
    int ts = dir ? (L_-1-t) : t;
    // gates = h @ Whh^T : M=16(8 valid) N=1024 K=256; 8 waves x 8 n-tiles,
    // K-outer so the A (h) fragment is loaded once per K-step.
    v8f acc[8];
#pragma unroll
    for (int i = 0; i < 8; ++i) acc[i] = (v8f){0,0,0,0,0,0,0,0};
    for (int k0 = 0; k0 < H_; k0 += 32){
      v16bf a = loadA_frag(hrow, k0, half);
#pragma unroll
      for (int i = 0; i < 8; ++i){
        int n0 = (wave*8 + i)*16;
        v16bf b = loadB_frag(Whh + (size_t)(n0+row)*H_, k0, half);
        acc[i] = __builtin_amdgcn_wmma_f32_16x16x32_bf16(false, a, false, b, (short)0, acc[i], false, false);
      }
    }
#pragma unroll
    for (int i = 0; i < 8; ++i){
      int n0 = (wave*8 + i)*16;
#pragma unroll
      for (int r = 0; r < 8; ++r){
        int m = r + 8*half;
        if (m < B_)
          g_s[m*G4_ + n0 + row] = acc[i][r] + Gx[((size_t)m*L_ + ts)*G4_ + n0 + row];
      }
    }
    __syncthreads();
    for (int idx = tid; idx < B_*H_; idx += 256){
      int m = idx >> 8, j = idx & (H_-1);
      const float* g = g_s + m*G4_;
      float ig = sigmoidf_(g[j]);
      float fg = sigmoidf_(g[H_ + j]);
      float gg = tanhf(g[2*H_ + j]);
      float og = sigmoidf_(g[3*H_ + j]);
      float cn = fg*c_s[idx] + ig*gg;
      float hn = og*tanhf(cn);
      float mt = mask[m*L_ + ts];
      float hv = mt*hn + (1.0f-mt)*h_s[idx];
      float cv = mt*cn + (1.0f-mt)*c_s[idx];
      h_s[idx] = hv; c_s[idx] = cv;
      hbf_s[m*H_ + j] = (bf16)hv;
      out[((size_t)m*L_ + ts)*H_ + j] = hv;
    }
    __syncthreads();
  }
}

// ----------------------------------------------------- Henc = [hf|hb]*mask
__global__ void concat_kernel(const float* __restrict__ hf, const float* __restrict__ hb,
                              const float* __restrict__ mask,
                              float* __restrict__ Hc, float* __restrict__ Hg, float* __restrict__ Hr){
  int i = blockIdx.x*blockDim.x + threadIdx.x;
  if (i >= B_*L_*H2_) return;
  int bl = i / H2_, j = i % H2_;
  float v = (j < H_) ? hf[(size_t)bl*H_ + j] : hb[(size_t)bl*H_ + (j-H_)];
  v *= mask[bl];
  Hc[i]=v; Hg[i]=v; Hr[i]=v;
}

// --------- BIO head (all 32 lanes active): mode 0 softmax, 1 log_softmax*mask
__global__ __launch_bounds__(32)
void bio_kernel(const float* __restrict__ Hg, const float* __restrict__ W_Cg,
                const float* __restrict__ b_Cg, const float* __restrict__ mask,
                float* __restrict__ probs, float* __restrict__ out, int mode){
  int bl = blockIdx.x;
  int lane = threadIdx.x;
  const float* h = Hg + (size_t)bl*H2_;
  float acc[NBIO_];
#pragma unroll
  for (int c = 0; c < NBIO_; ++c) acc[c] = 0.0f;
  for (int k = lane; k < H2_; k += 32){
    float hv = h[k];
#pragma unroll
    for (int c = 0; c < NBIO_; ++c) acc[c] += hv * W_Cg[(size_t)c*H2_ + k];
  }
#pragma unroll
  for (int c = 0; c < NBIO_; ++c)
    for (int off = 16; off >= 1; off >>= 1) acc[c] += __shfl_xor(acc[c], off);
  float logit[NBIO_];
  float m = -3.0e38f;
#pragma unroll
  for (int c = 0; c < NBIO_; ++c){ logit[c] = acc[c] + b_Cg[c]; m = fmaxf(m, logit[c]); }
  float sum = 0.0f;
#pragma unroll
  for (int c = 0; c < NBIO_; ++c) sum += expf(logit[c] - m);
  if (lane < NBIO_){
    if (mode == 0) probs[(size_t)bl*NBIO_ + lane] = expf(logit[lane] - m) / sum;
    else           out  [(size_t)bl*NBIO_ + lane] = (logit[lane] - m - logf(sum)) * mask[bl];
  }
}

// -------- fused pair scores: e=relu(a[s]+b2[t]+b_Lr) contracted with W_Cr^T
// K-outer, 8 t-tile accumulators; W_Cr fragment + a[s] loaded once per K-step.
// MODE 0: masked max over t -> pmax; MODE 1: full sigmoid*m3 -> out
template<int MODE>
__global__ __launch_bounds__(32)
void pair_kernel(const float* __restrict__ a_buf, const float* __restrict__ b2_buf,
                 const float* __restrict__ b_Lr, const bf16* __restrict__ WCr,
                 const float* __restrict__ b_Cr, const float* __restrict__ mask,
                 float* __restrict__ pmax, float* __restrict__ out){
  int bl = blockIdx.x;                 // b*L + s
  int b  = bl / L_;
  int lane = threadIdx.x, half = lane >> 4, row = lane & 15;
  __shared__ alignas(16) float asl[H2_];
  for (int k = lane; k < H2_; k += 32)
    asl[k] = a_buf[(size_t)bl*H2_ + k] + b_Lr[k];
  __syncthreads();
  bool svalid = mask[bl] > 0.0f;
  const bf16* wr = WCr + (size_t)row*H2_;              // this lane's label row
  v8f acc[8];
#pragma unroll
  for (int i = 0; i < 8; ++i) acc[i] = (v8f){0,0,0,0,0,0,0,0};
  for (int k0 = 0; k0 < H2_; k0 += 32){
    v16bf bw = loadB_frag(wr, k0, half);
    int ka0 = k0 + 8*half, ka1 = k0 + 16 + 8*half;     // A-fragment runs
    v4f s0 = *(const v4f*)(asl + ka0);
    v4f s1 = *(const v4f*)(asl + ka0 + 4);
    v4f s2 = *(const v4f*)(asl + ka1);
    v4f s3 = *(const v4f*)(asl + ka1 + 4);
#pragma unroll
    for (int i = 0; i < 8; ++i){
      int ta = i*16 + row;                             // this lane's e-row (t)
      const float* b2r = b2_buf + ((size_t)b*L_ + ta)*H2_;
      v4f c0 = *(const v4f*)(b2r + ka0);
      v4f c1 = *(const v4f*)(b2r + ka0 + 4);
      v4f c2 = *(const v4f*)(b2r + ka1);
      v4f c3 = *(const v4f*)(b2r + ka1 + 4);
      v16bf a;
#pragma unroll
      for (int q = 0; q < 4; ++q){
        a[q]    = (bf16)fmaxf(s0[q]+c0[q], 0.0f);
        a[4+q]  = (bf16)fmaxf(s1[q]+c1[q], 0.0f);
        a[8+q]  = (bf16)fmaxf(s2[q]+c2[q], 0.0f);
        a[12+q] = (bf16)fmaxf(s3[q]+c3[q], 0.0f);
      }
      acc[i] = __builtin_amdgcn_wmma_f32_16x16x32_bf16(false, a, false, bw, (short)0, acc[i], false, false);
    }
  }
  if (MODE == 0){
    float mx = -3.0e38f;
#pragma unroll
    for (int i = 0; i < 8; ++i)
#pragma unroll
      for (int r = 0; r < 8; ++r){
        int t = i*16 + r + 8*half;
        if (mask[b*L_ + t] > 0.0f) mx = fmaxf(mx, acc[i][r]);
      }
    mx = fmaxf(mx, __shfl_xor(mx, 16));                // fold t rows 0-7 / 8-15
    if (half == 0 && row < NLAB_){
      float v = (svalid && mx > -1.0e30f) ? sigmoidf_(mx + b_Cr[row]) : NEG_;
      pmax[(size_t)bl*NLAB_ + row] = v;
    }
  } else {
#pragma unroll
    for (int i = 0; i < 8; ++i)
#pragma unroll
      for (int r = 0; r < 8; ++r){
        int t = i*16 + r + 8*half;
        if (row < NLAB_){
          float m3 = (svalid && (mask[b*L_ + t] > 0.0f)) ? 1.0f : 0.0f;
          out[(((size_t)bl)*L_ + t)*NLAB_ + row] = m3 * sigmoidf_(acc[i][r] + b_Cr[row]);
        }
      }
  }
}

// -------------------------------------------------- GRU elementwise (gi tiny)
__global__ void gru_elem_kernel(const float* __restrict__ inp, int NI,
                                const float* __restrict__ Wih, const float* __restrict__ bih,
                                const float* __restrict__ GH, const float* __restrict__ Hc,
                                float* __restrict__ Hout){
  int i = blockIdx.x*blockDim.x + threadIdx.x;
  if (i >= B_*L_*H2_) return;
  int rowi = i / H2_, j = i % H2_;
  const float* xr = inp + (size_t)rowi*NI;
  float gr = bih[j], gz = bih[H2_+j], gn = bih[2*H2_+j];
  const float* wr = Wih + (size_t)j*NI;
  const float* wz = Wih + (size_t)(H2_+j)*NI;
  const float* wn = Wih + (size_t)(2*H2_+j)*NI;
  for (int c = 0; c < NI; ++c){
    float xv = xr[c];
    gr += xv*wr[c]; gz += xv*wz[c]; gn += xv*wn[c];
  }
  const float* ghr = GH + (size_t)rowi*G3_;      // bhh already added by GEMM
  float r = sigmoidf_(gr + ghr[j]);
  float z = sigmoidf_(gz + ghr[H2_+j]);
  float n = tanhf(gn + r*ghr[2*H2_+j]);
  float h = Hc[i];
  Hout[i] = (1.0f - z)*n + z*h;
}

__global__ void residual_kernel(float* __restrict__ Hc, const float* __restrict__ Hg,
                                const float* __restrict__ Hr){
  int i = blockIdx.x*blockDim.x + threadIdx.x;
  if (i < B_*L_*H2_) Hc[i] += Hg[i] + Hr[i];
}

// ===========================================================================
extern "C" void kernel_launch(void* const* d_in, const int* in_sizes, int n_in,
                              void* d_out, int out_size, void* d_ws, size_t ws_size,
                              hipStream_t stream){
  (void)in_sizes; (void)n_in; (void)out_size; (void)ws_size;
  const int*   tokens = (const int*)d_in[0];
  const int*   pos    = (const int*)d_in[1];
  const float* mask   = (const float*)d_in[2];
  const float* emb    = (const float*)d_in[3];
  const float* pemb   = (const float*)d_in[4];
  const float* Wih_f  = (const float*)d_in[5];
  const float* Whh_f  = (const float*)d_in[6];
  const float* b_f    = (const float*)d_in[7];
  const float* Wih_b  = (const float*)d_in[8];
  const float* Whh_b  = (const float*)d_in[9];
  const float* b_b    = (const float*)d_in[10];
  const float* Wg_ih  = (const float*)d_in[11];
  const float* Wg_hh  = (const float*)d_in[12];
  const float* bg_ih  = (const float*)d_in[13];
  const float* bg_hh  = (const float*)d_in[14];
  const float* Wr_ih  = (const float*)d_in[15];
  const float* Wr_hh  = (const float*)d_in[16];
  const float* br_ih  = (const float*)d_in[17];
  const float* br_hh  = (const float*)d_in[18];
  const float* W_Lr   = (const float*)d_in[19];
  const float* b_Lr   = (const float*)d_in[20];
  const float* W_Cr   = (const float*)d_in[21];
  const float* b_Cr   = (const float*)d_in[22];
  const float* W_Cg   = (const float*)d_in[23];
  const float* b_Cg   = (const float*)d_in[24];
  float* out_Cg = (float*)d_out;
  float* out_Cr = out_Cg + (size_t)B_*L_*NBIO_;

  // ---- workspace carve-up (all L2-resident, ~38 MB)
  char* p = (char*)d_ws;
  auto alloc = [&](size_t bytes)->char*{
    char* r = p; p += (bytes + 255) & ~((size_t)255); return r;
  };
  float* x        = (float*)alloc(sizeof(float)*B_*L_*IN_);
  bf16*  x_bf     = (bf16*) alloc(sizeof(bf16)*B_*L_*INP_);
  bf16*  Wihf_bf  = (bf16*) alloc(sizeof(bf16)*G4_*INP_);
  bf16*  Wihb_bf  = (bf16*) alloc(sizeof(bf16)*G4_*INP_);
  bf16*  Whhf_bf  = (bf16*) alloc(sizeof(bf16)*G4_*H_);
  bf16*  Whhb_bf  = (bf16*) alloc(sizeof(bf16)*G4_*H_);
  bf16*  Wghh_bf  = (bf16*) alloc(sizeof(bf16)*G3_*H2_);
  bf16*  Wrhh_bf  = (bf16*) alloc(sizeof(bf16)*G3_*H2_);
  bf16*  W1_bf    = (bf16*) alloc(sizeof(bf16)*H2_*H2_);
  bf16*  W2_bf    = (bf16*) alloc(sizeof(bf16)*H2_*H2_);
  bf16*  WCr_bf   = (bf16*) alloc(sizeof(bf16)*16*H2_);     // padded 10->16 rows
  float* Gx_f     = (float*)alloc(sizeof(float)*B_*L_*G4_);
  float* Gx_b     = (float*)alloc(sizeof(float)*B_*L_*G4_);
  float* hf       = (float*)alloc(sizeof(float)*B_*L_*H_);
  float* hb       = (float*)alloc(sizeof(float)*B_*L_*H_);
  float* Hc       = (float*)alloc(sizeof(float)*B_*L_*H2_);
  float* Hg       = (float*)alloc(sizeof(float)*B_*L_*H2_);
  float* Hr       = (float*)alloc(sizeof(float)*B_*L_*H2_);
  bf16*  Hc_bf    = (bf16*) alloc(sizeof(bf16)*B_*L_*H2_);
  bf16*  Hr_bf    = (bf16*) alloc(sizeof(bf16)*B_*L_*H2_);
  float* a_buf    = (float*)alloc(sizeof(float)*B_*L_*H2_);
  float* b2_buf   = (float*)alloc(sizeof(float)*B_*L_*H2_);
  float* prob_Cg  = (float*)alloc(sizeof(float)*B_*L_*NBIO_);
  float* prob_Crm = (float*)alloc(sizeof(float)*B_*L_*NLAB_);
  float* GH       = (float*)alloc(sizeof(float)*B_*L_*G3_);

  const int BL = B_*L_;
  auto packN = [&](bf16* dst, const float* src, int dr, int dc, int sr, int sc,
                   int stride, int off){
    int n = dr*dc;
    pack_bf16_kernel<<<(n+255)/256, 256, 0, stream>>>(dst, src, dr, dc, sr, sc, stride, off);
  };
  auto gemm = [&](const bf16* A, int lda, const bf16* W, int ldw, const float* bias,
                  float* C, int ldc, int M, int N, int K){
    gemm_wmma_kernel<<<dim3(N/32, M/32), 32, 0, stream>>>(A, lda, W, ldw, bias, C, ldc, K);
  };

  // ---- weight packs (bf16, K zero-padded to x32)
  packN(Wihf_bf, Wih_f, G4_, INP_, G4_, IN_, IN_, 0);
  packN(Wihb_bf, Wih_b, G4_, INP_, G4_, IN_, IN_, 0);
  packN(Whhf_bf, Whh_f, G4_, H_,  G4_, H_,  H_,  0);
  packN(Whhb_bf, Whh_b, G4_, H_,  G4_, H_,  H_,  0);
  packN(Wghh_bf, Wg_hh, G3_, H2_, G3_, H2_, H2_, 0);
  packN(Wrhh_bf, Wr_hh, G3_, H2_, G3_, H2_, H2_, 0);
  packN(W1_bf,   W_Lr,  H2_, H2_, H2_, H2_, 2*H2_, 0);     // W_Lr[:, :H2]
  packN(W2_bf,   W_Lr,  H2_, H2_, H2_, H2_, 2*H2_, H2_);   // W_Lr[:, H2:]
  packN(WCr_bf,  W_Cr,  16,  H2_, NLAB_, H2_, H2_, 0);     // zero-padded labels

  // ---- embedding + hoisted LSTM input GEMMs
  embed_kernel<<<BL, 128, 0, stream>>>(tokens, pos, emb, pemb, x);
  packN(x_bf, x, BL, INP_, BL, IN_, IN_, 0);
  gemm(x_bf, INP_, Wihf_bf, INP_, b_f, Gx_f, G4_, BL, G4_, INP_);
  gemm(x_bf, INP_, Wihb_bf, INP_, b_b, Gx_b, G4_, BL, G4_, INP_);

  // ---- bidirectional recurrence: one persistent block per direction
  lstm_kernel<<<2, 256, 0, stream>>>(Gx_f, Gx_b, Whhf_bf, Whhb_bf, mask, hf, hb);

  concat_kernel<<<(BL*H2_+255)/256, 256, 0, stream>>>(hf, hb, mask, Hc, Hg, Hr);

  // ---- refinement rounds
  for (int round = 0; round < 2; ++round){
    // heads from current Hg / Hr (before GRU updates)
    bio_kernel<<<BL, 32, 0, stream>>>(Hg, W_Cg, b_Cg, mask, prob_Cg, nullptr, 0);
    packN(Hr_bf, Hr, BL, H2_, BL, H2_, H2_, 0);
    gemm(Hr_bf, H2_, W1_bf, H2_, nullptr, a_buf,  H2_, BL, H2_, H2_);
    gemm(Hr_bf, H2_, W2_bf, H2_, nullptr, b2_buf, H2_, BL, H2_, H2_);
    pair_kernel<0><<<BL, 32, 0, stream>>>(a_buf, b2_buf, b_Lr, WCr_bf, b_Cr, mask,
                                          prob_Crm, nullptr);
    // GRU updates, both off the same Hc
    packN(Hc_bf, Hc, BL, H2_, BL, H2_, H2_, 0);
    gemm(Hc_bf, H2_, Wghh_bf, H2_, bg_hh, GH, G3_, BL, G3_, H2_);
    gru_elem_kernel<<<(BL*H2_+255)/256, 256, 0, stream>>>(prob_Cg, NBIO_, Wg_ih, bg_ih, GH, Hc, Hg);
    gemm(Hc_bf, H2_, Wrhh_bf, H2_, br_hh, GH, G3_, BL, G3_, H2_);
    gru_elem_kernel<<<(BL*H2_+255)/256, 256, 0, stream>>>(prob_Crm, NLAB_, Wr_ih, br_ih, GH, Hc, Hr);
    residual_kernel<<<(BL*H2_+255)/256, 256, 0, stream>>>(Hc, Hg, Hr);
  }

  // ---- final outputs
  bio_kernel<<<BL, 32, 0, stream>>>(Hg, W_Cg, b_Cg, mask, nullptr, out_Cg, 1);
  packN(Hr_bf, Hr, BL, H2_, BL, H2_, H2_, 0);
  gemm(Hr_bf, H2_, W1_bf, H2_, nullptr, a_buf,  H2_, BL, H2_, H2_);
  gemm(Hr_bf, H2_, W2_bf, H2_, nullptr, b2_buf, H2_, BL, H2_, H2_);
  pair_kernel<1><<<BL, 32, 0, stream>>>(a_buf, b2_buf, b_Lr, WCr_bf, b_Cr, mask,
                                        nullptr, out_Cr);
}
